// Attention_57930518889160
// MI455X (gfx1250) — matrix operand order repaired and checked by reading, not compiled, not run on hardware
//
#include <hip/hip_runtime.h>
#include <stdint.h>

#define DIM    4096
#define KVDIM  1024
#define NHEADS 32
#define NREP   4
#define HD     128
#define BSZ    2
#define SEQ    2048
#define MROWS  (BSZ * SEQ)   // 4096 token rows

typedef __bf16 bf16;
typedef __attribute__((ext_vector_type(16))) __bf16    v16bf;
typedef __attribute__((ext_vector_type(8)))  float     v8f;
typedef __attribute__((ext_vector_type(4)))  uint32_t  u32x4;
typedef __attribute__((ext_vector_type(8)))  uint32_t  u32x8;

union Frag { v16bf v; uint4 q[2]; };
union Oct  { uint4 q; bf16 h[8]; };

__device__ __forceinline__ bf16 f2bf(float f) {
  union { float f; uint32_t u; } in; in.f = f;
  uint32_t r = (in.u + 0x7FFFu + ((in.u >> 16) & 1u)) >> 16;   // RNE
  union { uint16_t s; bf16 b; } out; out.s = (uint16_t)r;
  return out.b;
}

// ---- CDNA5 async global->LDS (bypasses VGPRs, tracked by ASYNCcnt) ----------
__device__ __forceinline__ void async_load_b128(const void* gptr, void* lptr) {
  uint32_t lds = (uint32_t)(uintptr_t)lptr;   // low 32 bits of flat addr = LDS offset
  asm volatile("global_load_async_to_lds_b128 %0, %1, off"
               :: "v"(lds), "v"(gptr) : "memory");
}
__device__ __forceinline__ void wait_async0() {
  asm volatile("s_wait_asynccnt 0" ::: "memory");
}

// ---- CDNA5 Tensor Data Mover: 2D tile [64 keys x 128 dims] of bf16 to LDS ---
// LDS rows padded: 128 elems (64 DWORDs) then +4 DWORDs -> 136-elem row stride.
__device__ __forceinline__ void tdm_load_k_tile(const bf16* gsrc, uint32_t lds_byte) {
  uint64_t ga = (uint64_t)(uintptr_t)gsrc;
  u32x4 g0;
  g0[0] = 1u;                                            // count=1 user D#
  g0[1] = lds_byte;                                      // lds_addr
  g0[2] = (uint32_t)(ga & 0xFFFFFFFFu);                  // global_addr[31:0]
  g0[3] = (uint32_t)((ga >> 32) & 0x1FFFFFFu) | (2u << 30); // addr[56:32], type=2
  u32x8 g1;
  g1[0] = (1u << 16)        // data_size = 2 bytes
        | (1u << 20)        // pad_enable
        | (5u << 22)        // pad_interval: 64 DWORDs (256B) before padding
        | (3u << 25);       // pad_amount: 4 DWORDs (16B)
  g1[1] = (128u << 16);     // tensor_dim0[15:0] = 128 (bits 63:48)
  g1[2] = (64u << 16);      // tensor_dim1[15:0] = 64  (bits 95:80)
  g1[3] = (128u << 16);     // tile_dim0 = 128 (bits 127:112)
  g1[4] = 64u;              // tile_dim1 = 64  (bits 143:128), tile_dim2 = 0
  g1[5] = (uint32_t)KVDIM;  // tensor_dim0_stride[31:0] = 1024 elems
  g1[6] = 0u;               // stride0[47:32]=0, tensor_dim1_stride lo = 0
  g1[7] = 0u;
  asm volatile("tensor_load_to_lds %0, %1" :: "s"(g0), "s"(g1) : "memory");
}

// ---------------------------------------------------------------- fp32 -> bf16
__global__ void f32_to_bf16_kernel(const float* __restrict__ in,
                                   bf16* __restrict__ out, int n) {
  int i = blockIdx.x * blockDim.x + threadIdx.x;
  if (i < n) out[i] = f2bf(in[i]);
}

// ---------------------------------------------------------------- GEMM
// C[M,N] = A[M,K=DIM] @ W[N,K=DIM]^T + bias
// MODE 0: bf16 out (V proj) | MODE 1: bf16 out + RoPE (Q/K) | MODE 2: f32 out
template <int MODE>
__global__ __launch_bounds__(256)
void gemm_bf16_kernel(const bf16* __restrict__ A,
                      const bf16* __restrict__ W,
                      const float* __restrict__ bias,
                      const float* __restrict__ fcos,
                      const float* __restrict__ fsin,
                      bf16* __restrict__ Cb,
                      float* __restrict__ Cf,
                      int N) {
  __shared__ alignas(16) bf16 As[2 * 128 * 40];   // double-buffered 128x32 tiles
  __shared__ alignas(16) bf16 Bs[2 * 128 * 40];

  const int tid  = threadIdx.x;
  const int lane = tid & 31;
  const int wave = tid >> 5;
  const int hl   = lane >> 4;       // lane half (wave32)
  const int l16  = lane & 15;
  const int m0 = blockIdx.y * 128;
  const int n0 = blockIdx.x * 128;
  const int wm = (wave >> 2) * 64;  // wave tile: 64(M) x 32(N)
  const int wn = (wave & 3) * 32;

  v8f acc[4][2];
  #pragma unroll
  for (int i = 0; i < 4; ++i)
    #pragma unroll
    for (int j = 0; j < 2; ++j)
      #pragma unroll
      for (int r = 0; r < 8; ++r) acc[i][j][r] = 0.f;

  auto load_tile = [&](int kk, int buf) {      // 4 async b128 per thread
    #pragma unroll
    for (int it = 0; it < 2; ++it) {
      int o = tid + it * 256;
      int row = o >> 2, c8 = (o & 3) * 8;
      async_load_b128(&A[(size_t)(m0 + row) * DIM + kk + c8],
                      &As[buf * (128 * 40) + row * 40 + c8]);
      async_load_b128(&W[(size_t)(n0 + row) * DIM + kk + c8],
                      &Bs[buf * (128 * 40) + row * 40 + c8]);
    }
  };

  load_tile(0, 0);
  wait_async0();
  __syncthreads();

  int cur = 0;
  for (int kk = 0; kk < DIM; kk += 32) {
    if (kk + 32 < DIM) load_tile(kk + 32, cur ^ 1);   // DMA next tile under WMMA

    const bf16* Ac = &As[cur * (128 * 40)];
    const bf16* Bc = &Bs[cur * (128 * 40)];
    Frag af[4], bfr[2];
    #pragma unroll
    for (int i = 0; i < 4; ++i) {               // A 16x32: half -> K 0..7 / 8..15, +16
      int am = wm + i * 16 + l16;
      int ko = hl * 8;
      af[i].q[0] = *(const uint4*)&Ac[am * 40 + ko];
      af[i].q[1] = *(const uint4*)&Ac[am * 40 + ko + 16];
    }
    #pragma unroll
    for (int j = 0; j < 2; ++j) {               // B 32x16: half -> K 0..7 / 8..15, +16
      int bn = wn + j * 16 + l16;
      int ko = hl * 8;
      bfr[j].q[0] = *(const uint4*)&Bc[bn * 40 + ko];
      bfr[j].q[1] = *(const uint4*)&Bc[bn * 40 + ko + 16];
    }
    #pragma unroll
    for (int i = 0; i < 4; ++i)
      #pragma unroll
      for (int j = 0; j < 2; ++j)
        acc[i][j] = __builtin_amdgcn_wmma_f32_16x16x32_bf16(
            false, af[i].v, false, bfr[j].v, (short)0, acc[i][j], false, false);

    wait_async0();
    __syncthreads();
    cur ^= 1;
  }

  // epilogue: bias (+ optional RoPE via adjacent-lane exchange) + store
  #pragma unroll
  for (int i = 0; i < 4; ++i) {
    #pragma unroll
    for (int j = 0; j < 2; ++j) {
      const int n = n0 + wn + j * 16 + l16;
      const float bn = bias[n];
      #pragma unroll
      for (int r = 0; r < 8; ++r) {
        const int m = m0 + wm + i * 16 + r + 8 * hl;
        float val = acc[i][j][r] + bn;
        if (MODE == 1) {                         // fused RoPE (pairs = adjacent lanes)
          int srow = m & (SEQ - 1);
          int jj = (n & (HD - 1)) >> 1;
          float c = fcos[srow * (HD / 2) + jj];
          float s = fsin[srow * (HD / 2) + jj];
          float other = __shfl_xor(val, 1, 32);
          val = (n & 1) ? (val * c + other * s) : (val * c - other * s);
        }
        if (MODE == 2) Cf[(size_t)m * N + n] = val;
        else           Cb[(size_t)m * N + n] = f2bf(val);
      }
    }
  }
}

// ---------------------------------------------------------------- flash attention
// grid.x = BSZ*NHEADS, grid.y = SEQ/128.  Each wave: 16 queries x full head_dim.
__global__ __launch_bounds__(256)
void flash_attn_kernel(const bf16* __restrict__ Q,   // [MROWS][DIM]
                       const bf16* __restrict__ K,   // [MROWS][KVDIM]
                       const bf16* __restrict__ V,   // [MROWS][KVDIM]
                       bf16* __restrict__ O) {       // [MROWS][DIM]
  __shared__ alignas(16) bf16 Ks[64 * 136];          // keys row-major (TDM-padded rows)
  __shared__ alignas(16) bf16 Vt[128 * 72];          // V transposed [dim][key]
  __shared__ alignas(16) bf16 Ps[8 * 16 * 72];       // per-wave P tiles

  const int tid  = threadIdx.x;
  const int lane = tid & 31;
  const int wave = tid >> 5;
  const int hl   = lane >> 4;
  const int l16  = lane & 15;
  const int b = blockIdx.x / NHEADS;
  const int h = blockIdx.x % NHEADS;
  const int g = h / NREP;                            // kv head
  const int q0 = blockIdx.y * 128;

  // Q fragments for this wave's 16 rows stay resident (32 VGPRs)
  const size_t qrow = (size_t)(b * SEQ + q0 + wave * 16 + l16);
  Frag qf[4];
  #pragma unroll
  for (int d = 0; d < 4; ++d) {
    int ko = d * 32 + hl * 8;
    qf[d].q[0] = *(const uint4*)&Q[qrow * DIM + h * HD + ko];
    qf[d].q[1] = *(const uint4*)&Q[qrow * DIM + h * HD + ko + 16];
  }

  v8f oacc[8];
  #pragma unroll
  for (int dt = 0; dt < 8; ++dt)
    #pragma unroll
    for (int r = 0; r < 8; ++r) oacc[dt][r] = 0.f;
  float mrow[8], lrow[8];
  #pragma unroll
  for (int r = 0; r < 8; ++r) { mrow[r] = -3.0e38f; lrow[r] = 0.f; }

  const float scale = 0.08838834764831845f;          // 1/sqrt(128)
  const int kbMax = q0 + 128;                        // causal: skip fully-masked blocks

  for (int kb = 0; kb < kbMax; kb += 64) {
    // K tile [64 x 128] via Tensor Data Mover (one issuing wave)
    if (wave == 0)
      tdm_load_k_tile(&K[(size_t)(b * SEQ + kb) * KVDIM + g * HD],
                      (uint32_t)(uintptr_t)&Ks[0]);
    // V tile: per-lane load + transposed scatter into Vt
    #pragma unroll
    for (int it = 0; it < 4; ++it) {
      int o = tid + it * 256;                        // 1024 octets
      int row = o >> 4;                              // key 0..63
      int c8 = (o & 15) * 8;                         // dim
      const size_t krow = (size_t)(b * SEQ + kb + row);
      Oct vv; vv.q = *(const uint4*)&V[krow * KVDIM + g * HD + c8];
      #pragma unroll
      for (int e = 0; e < 8; ++e) Vt[(c8 + e) * 72 + row] = vv.h[e];
    }
    __builtin_amdgcn_s_wait_tensorcnt(0);            // no-op for non-issuing waves
    __syncthreads();

    // S(16x64) = Q @ K^T
    v8f st[4];
    #pragma unroll
    for (int nt = 0; nt < 4; ++nt)
      #pragma unroll
      for (int r = 0; r < 8; ++r) st[nt][r] = 0.f;
    #pragma unroll
    for (int d = 0; d < 4; ++d) {
      #pragma unroll
      for (int nt = 0; nt < 4; ++nt) {
        Frag kf;
        int kn = nt * 16 + l16;
        int ko = d * 32 + hl * 8;
        kf.q[0] = *(const uint4*)&Ks[kn * 136 + ko];
        kf.q[1] = *(const uint4*)&Ks[kn * 136 + ko + 16];
        st[nt] = __builtin_amdgcn_wmma_f32_16x16x32_bf16(
            false, qf[d].v, false, kf.v, (short)0, st[nt], false, false);
      }
    }

    // scale + causal mask
    #pragma unroll
    for (int nt = 0; nt < 4; ++nt)
      #pragma unroll
      for (int r = 0; r < 8; ++r) {
        int qp = q0 + wave * 16 + r + 8 * hl;
        int kp = kb + nt * 16 + l16;
        float v = st[nt][r] * scale;
        st[nt][r] = (kp > qp) ? -1.0e30f : v;
      }

    // online softmax (row = half-wave shfl reduction)
    #pragma unroll
    for (int r = 0; r < 8; ++r) {
      float mx = fmaxf(fmaxf(st[0][r], st[1][r]), fmaxf(st[2][r], st[3][r]));
      mx = fmaxf(mx, __shfl_xor(mx, 1, 32));
      mx = fmaxf(mx, __shfl_xor(mx, 2, 32));
      mx = fmaxf(mx, __shfl_xor(mx, 4, 32));
      mx = fmaxf(mx, __shfl_xor(mx, 8, 32));
      float mnew  = fmaxf(mrow[r], mx);
      float alpha = __expf(mrow[r] - mnew);
      mrow[r] = mnew;
      float rs = 0.f;
      #pragma unroll
      for (int nt = 0; nt < 4; ++nt) {
        float p = __expf(st[nt][r] - mnew);
        st[nt][r] = p;
        rs += p;
      }
      rs += __shfl_xor(rs, 1, 32);
      rs += __shfl_xor(rs, 2, 32);
      rs += __shfl_xor(rs, 4, 32);
      rs += __shfl_xor(rs, 8, 32);
      lrow[r] = lrow[r] * alpha + rs;
      #pragma unroll
      for (int dt = 0; dt < 8; ++dt) oacc[dt][r] *= alpha;
      int ml = r + 8 * hl;                          // stage P for re-fragmenting
      #pragma unroll
      for (int nt = 0; nt < 4; ++nt)
        Ps[wave * 1152 + ml * 72 + nt * 16 + l16] = f2bf(st[nt][r]);
    }

    asm volatile("s_wait_dscnt 0" ::: "memory");    // wave-private P RAW through LDS

    // O(16x128) += P(16x64) @ V(64x128)
    #pragma unroll
    for (int kc = 0; kc < 2; ++kc) {
      Frag pf;
      int ko = kc * 32 + hl * 8;
      pf.q[0] = *(const uint4*)&Ps[wave * 1152 + l16 * 72 + ko];
      pf.q[1] = *(const uint4*)&Ps[wave * 1152 + l16 * 72 + ko + 16];
      #pragma unroll
      for (int dt = 0; dt < 8; ++dt) {
        Frag vf;
        int vn = dt * 16 + l16;
        int vo = kc * 32 + hl * 8;
        vf.q[0] = *(const uint4*)&Vt[vn * 72 + vo];
        vf.q[1] = *(const uint4*)&Vt[vn * 72 + vo + 16];
        oacc[dt] = __builtin_amdgcn_wmma_f32_16x16x32_bf16(
            false, pf.v, false, vf.v, (short)0, oacc[dt], false, false);
      }
    }
    __syncthreads();
  }

  #pragma unroll
  for (int dt = 0; dt < 8; ++dt)
    #pragma unroll
    for (int r = 0; r < 8; ++r) {
      int qp = q0 + wave * 16 + r + 8 * hl;
      size_t row = (size_t)(b * SEQ + qp);
      O[row * DIM + h * HD + dt * 16 + l16] = f2bf(oacc[dt][r] / lrow[r]);
    }
}

// ---------------------------------------------------------------- host
extern "C" void kernel_launch(void* const* d_in, const int* in_sizes, int n_in,
                              void* d_out, int out_size, void* d_ws, size_t ws_size,
                              hipStream_t stream) {
  const float* x    = (const float*)d_in[0];
  const float* fcos = (const float*)d_in[1];
  const float* fsin = (const float*)d_in[2];
  // d_in[3] = mask: replaced by inline causal logic
  const float* wq = (const float*)d_in[4];
  const float* bq = (const float*)d_in[5];
  const float* wk = (const float*)d_in[6];
  const float* bk = (const float*)d_in[7];
  const float* wv = (const float*)d_in[8];
  const float* bv = (const float*)d_in[9];
  const float* wo = (const float*)d_in[10];
  const float* bo = (const float*)d_in[11];
  float* out = (float*)d_out;

  char* ws = (char*)d_ws;
  size_t off = 0;
  auto alloc = [&](size_t elems) {
    bf16* p = (bf16*)(ws + off);
    off += elems * sizeof(bf16);
    return p;
  };
  bf16* xb  = alloc((size_t)MROWS * DIM);
  bf16* wqb = alloc((size_t)DIM * DIM);
  bf16* wkb = alloc((size_t)KVDIM * DIM);
  bf16* wvb = alloc((size_t)KVDIM * DIM);
  bf16* wob = alloc((size_t)DIM * DIM);
  bf16* Qb  = alloc((size_t)MROWS * DIM);
  bf16* Kb  = alloc((size_t)MROWS * KVDIM);
  bf16* Vb  = alloc((size_t)MROWS * KVDIM);
  bf16* Ab  = alloc((size_t)MROWS * DIM);
  (void)ws_size; (void)n_in; (void)in_sizes; (void)out_size;

  auto cvt = [&](const float* src, bf16* dst, int n) {
    f32_to_bf16_kernel<<<dim3((unsigned)((n + 255) / 256)), dim3(256), 0, stream>>>(src, dst, n);
  };
  cvt(x,  xb,  MROWS * DIM);
  cvt(wq, wqb, DIM * DIM);
  cvt(wk, wkb, KVDIM * DIM);
  cvt(wv, wvb, KVDIM * DIM);
  cvt(wo, wob, DIM * DIM);

  gemm_bf16_kernel<1><<<dim3(DIM / 128,   MROWS / 128), 256, 0, stream>>>(xb, wqb, bq, fcos, fsin, Qb, nullptr, DIM);
  gemm_bf16_kernel<1><<<dim3(KVDIM / 128, MROWS / 128), 256, 0, stream>>>(xb, wkb, bk, fcos, fsin, Kb, nullptr, KVDIM);
  gemm_bf16_kernel<0><<<dim3(KVDIM / 128, MROWS / 128), 256, 0, stream>>>(xb, wvb, bv, nullptr, nullptr, Vb, nullptr, KVDIM);

  flash_attn_kernel<<<dim3(BSZ * NHEADS, SEQ / 128), 256, 0, stream>>>(Qb, Kb, Vb, Ab);

  gemm_bf16_kernel<2><<<dim3(DIM / 128, MROWS / 128), 256, 0, stream>>>(Ab, wob, bo, nullptr, nullptr, nullptr, out, DIM);
}
